// SSTFRLayer_687194768007
// MI455X (gfx1250) — matrix operand order repaired and checked by reading, not compiled, not run on hardware
//
#include <hip/hip_runtime.h>
#include <hip/hip_bf16.h>

// ---------------------------------------------------------------------------
// SSTFR truncated-FIR filterbank as f16 WMMA Toeplitz GEMM for gfx1250.
//   H[b,t,n] = sum_tau x[b, t - tau] * g[tau, n],  n in [0,256): re block, im block.
// K = 1152 (static from reference math), chunked as 36 x (16x16x32) WMMA steps.
// Each wave register-blocks 2 time tiles (32 t) x 1 (re,im) col pair (16 d),
// sharing B fragments across the two M tiles and overlapping A-fragment LDS
// loads (3 ds_load_b128 feed both A fragments per chunk).
// ---------------------------------------------------------------------------

typedef __attribute__((ext_vector_type(16))) _Float16 v16h;
typedef __attribute__((ext_vector_type(8)))  _Float16 v8h;
typedef __attribute__((ext_vector_type(8)))  float    v8f;

#define BATCH   8
#define LLEN    80000
#define DCH     128
#define KTAPS   1152
#define KC      (KTAPS / 32)          // 36 k-chunks
#define NT      16                    // 256 cols / 16
#define TBLK    32                    // times per workgroup (2 M tiles per wave)
#define XWIN    (KTAPS + TBLK)        // 1184 staged x values per tile
#define CPY     (XWIN + 8)            // 1192 halves per phase copy (2384 B, 16B mult)
#define GFRAG_ELEMS (KC * NT * 32 * 16)   // 294912 f16 = 576 KB workspace

// ---------------------------------------------------------------------------
// Kernel 1: build g in WMMA B-fragment layout.
// flat idx = ((kb*16 + nt)*32 + lane)*16 + j
//   column n   = nt*16 + (lane & 15)          (re if n<128 else im, d = n & 127)
//   row    tau = kb*32 + (j>>3)*16 + (lane>>4)*8 + (j&7)
// g[tau, n] = exp(b_log_mag[d] + tau*alpha[d]) * {cos,sin}(b_phase[d] + tau*omega[d])
// with alpha = -softplus(alpha_raw).
// ---------------------------------------------------------------------------
__global__ __launch_bounds__(256) void sstfr_gfrag_kernel(
    const float* __restrict__ omega, const float* __restrict__ alpha_raw,
    const float* __restrict__ b_log_mag, const float* __restrict__ b_phase,
    _Float16* __restrict__ gfrag) {
  int idx = blockIdx.x * 256 + threadIdx.x;
  if (idx >= GFRAG_ELEMS) return;
  int j    = idx & 15;
  int lane = (idx >> 4) & 31;
  int nt   = (idx >> 9) & 15;
  int kb   = idx >> 13;
  int n    = nt * 16 + (lane & 15);
  int tau  = kb * 32 + ((j >> 3) << 4) + ((lane >> 4) << 3) + (j & 7);
  int d    = n & (DCH - 1);
  float alpha = -log1pf(__expf(alpha_raw[d]));        // -softplus
  float ft    = (float)tau;
  float mag   = __expf(b_log_mag[d] + ft * alpha);
  float ph    = b_phase[d] + ft * omega[d];
  float val   = mag * ((n >> 7) ? __sinf(ph) : __cosf(ph));
  gfrag[idx] = (_Float16)val;
}

// ---------------------------------------------------------------------------
// Kernel 2: per workgroup (256 thr = 8 waves) compute a 32-time x 256-col tile.
// Wave w owns cols [16w,16w+16) for both re (tile w) and im (tile w+8), and
// two time tiles: lo = [t0, t0+16), hi = [t0+16, t0+32).
// x window staged reversed (xrev[i] = x[t0+31-i]) in 8 phase-shifted copies so
// every lane's 8-half A run is 16B aligned -> pure ds_load_b128.
//   A_hi[M,K] = xrev[15 - M + kb*32 + K],  A_lo[M,K] = xrev[31 - M + kb*32 + K]
//   -> hi runs at [off, off+32), [off+32..]; lo runs at [off+32..), [off+64..):
//      3 ds_load_b128 per chunk feed both fragments.
// ---------------------------------------------------------------------------
__global__ __launch_bounds__(256) void sstfr_wmma_kernel(
    const float* __restrict__ x, const _Float16* __restrict__ gfrag,
    float* __restrict__ out) {
  __shared__ __attribute__((aligned(16))) _Float16 smem[8 * CPY];  // 19072 B

  const int b   = blockIdx.y;
  const int t0  = blockIdx.x * TBLK;
  const int tid = threadIdx.x;
  const float* xb = x + (size_t)b * LLEN;

  // Stage: copy p, element i holds xrev[i-p] = x[b, t0+31-(i-p)] (0 if OOB).
  for (int s = tid; s < 8 * CPY; s += 256) {
    int p = s / CPY;
    int q = s - p * CPY - p;            // index into xrev
    float v = 0.0f;
    if (q >= 0 && q < XWIN) {
      int t = t0 + (TBLK - 1) - q;
      if (t >= 0) v = xb[t];
    }
    smem[s] = (_Float16)v;
  }
  __syncthreads();

  const int lane = tid & 31;            // wave32 lane
  const int w    = tid >> 5;            // wave id 0..7

  // A-fragment addressing (ISA 16-bit A 16x32 layout):
  //   halves 0..7  -> K = kkstart + j,  halves 8..15 -> K = 16 + kkstart + j
  const int M   = lane & 15;
  const int kk  = (lane >> 4) << 3;     // 0 or 8
  const int v10 = 15 - M + kk;          // xrev index of hi-tile half 0 at kb=0
  const int phz = (8 - (v10 & 7)) & 7;  // phase copy giving 16B alignment
  int aoff = phz * (CPY * 2) + 2 * (v10 + phz);   // byte offset, 16B aligned
  const char* sbase = (const char*)smem;

  // B-fragment pointers (v16h = one 32B fragment per lane per tile per chunk).
  const v16h* gre = (const v16h*)gfrag + (size_t)w * 32 + lane;
  const v16h* gim = gre + 8 * 32;       // im tile = w + 8
  v8f accRhi = {}, accIhi = {};
  v8f accRlo = {}, accIlo = {};

#pragma unroll 2
  for (int kb = 0; kb < KC; ++kb) {
    v8h l0 = *(const v8h*)(sbase + aoff);        // hi halves 0-7
    v8h l1 = *(const v8h*)(sbase + aoff + 32);   // hi halves 8-15 / lo halves 0-7
    v8h l2 = *(const v8h*)(sbase + aoff + 64);   // lo halves 8-15
    v16h ahi = __builtin_shufflevector(l0, l1,
                                       0, 1, 2, 3, 4, 5, 6, 7,
                                       8, 9, 10, 11, 12, 13, 14, 15);
    v16h alo = __builtin_shufflevector(l1, l2,
                                       0, 1, 2, 3, 4, 5, 6, 7,
                                       8, 9, 10, 11, 12, 13, 14, 15);
    v16h bR = gre[(size_t)kb * (NT * 32)];
    v16h bI = gim[(size_t)kb * (NT * 32)];
    accRhi = __builtin_amdgcn_wmma_f32_16x16x32_f16(
        false, ahi, false, bR, (short)0, accRhi, false, false);
    accIhi = __builtin_amdgcn_wmma_f32_16x16x32_f16(
        false, ahi, false, bI, (short)0, accIhi, false, false);
    accRlo = __builtin_amdgcn_wmma_f32_16x16x32_f16(
        false, alo, false, bR, (short)0, accRlo, false, false);
    accIlo = __builtin_amdgcn_wmma_f32_16x16x32_f16(
        false, alo, false, bI, (short)0, accIlo, false, false);
    aoff += 64;                          // advance 32 taps in xrev
  }

  // C/D layout: VGPR v, lanes 0-15 -> M=v, lanes 16-31 -> M=8+v; N = lane&15.
  // Output (B,L,D,2): interleave re/im as float2 -> contiguous 128B per 16 lanes.
  const int dcol  = (w << 4) + (lane & 15);
  const int mbase = (lane >> 4) << 3;
  float* oplo = out + (((size_t)b * LLEN + t0 + mbase) * DCH + dcol) * 2;
  float* ophi = oplo + (size_t)16 * DCH * 2;
#pragma unroll
  for (int v = 0; v < 8; ++v) {
    float2 vlo = make_float2(accRlo[v], accIlo[v]);
    float2 vhi = make_float2(accRhi[v], accIhi[v]);
    *(float2*)(oplo + (size_t)v * (DCH * 2)) = vlo;   // t = t0 + mbase + v
    *(float2*)(ophi + (size_t)v * (DCH * 2)) = vhi;   // t = t0 + 16 + mbase + v
  }
}

// ---------------------------------------------------------------------------
// Host launcher. Inputs: x, omega, alpha_raw, b_log_mag, b_phase, K(device,
// statically 1152). Workspace holds the 576 KB f16 B-fragment tensor.
// ---------------------------------------------------------------------------
extern "C" void kernel_launch(void* const* d_in, const int* in_sizes, int n_in,
                              void* d_out, int out_size, void* d_ws, size_t ws_size,
                              hipStream_t stream) {
  const float* x         = (const float*)d_in[0];
  const float* omega     = (const float*)d_in[1];
  const float* alpha_raw = (const float*)d_in[2];
  const float* b_log_mag = (const float*)d_in[3];
  const float* b_phase   = (const float*)d_in[4];
  (void)in_sizes; (void)n_in; (void)out_size; (void)ws_size;

  _Float16* gfrag = (_Float16*)d_ws;     // 294912 f16 = 576 KB
  float* out = (float*)d_out;

  int gblocks = (GFRAG_ELEMS + 255) / 256;
  sstfr_gfrag_kernel<<<dim3(gblocks), dim3(256), 0, stream>>>(
      omega, alpha_raw, b_log_mag, b_phase, gfrag);

  dim3 grid(LLEN / TBLK, BATCH);
  sstfr_wmma_kernel<<<grid, dim3(256), 0, stream>>>(x, gfrag, out);
}
